// SwinTransformerBlock_31696858644754
// MI455X (gfx1250) — matrix operand order repaired and checked by reading, compiled, not run on hardware
//
#include <hip/hip_runtime.h>

// ---------------------------------------------------------------------------
// Swin-3D block for MI455X (gfx1250, wave32, WMMA 16x16x32 f16/f32-acc)
// ---------------------------------------------------------------------------
typedef _Float16 h16;
typedef __attribute__((ext_vector_type(16))) _Float16 v16h;
typedef __attribute__((ext_vector_type(8)))  float    v8f;
typedef unsigned int u32x4 __attribute__((ext_vector_type(4)));
typedef int i32x4 __attribute__((ext_vector_type(4)));
typedef int i32x8 __attribute__((ext_vector_type(8)));

#define DEVINL static __device__ __forceinline__

#if __has_builtin(__builtin_amdgcn_tensor_load_to_lds) && \
    __has_builtin(__builtin_amdgcn_s_wait_tensorcnt)
#define USE_TDM 1
#else
#define USE_TDM 0
#endif

union HVec { uint32_t u[8]; v16h v; uint4 q[2]; };

// A-fragment (16x32 f16): lane m = lane&15, g = lane>>4.
// VGPR v holds packed pair k = (v>>2)*16 + g*8 + (v&3)*2  -> two contiguous
// 16-byte chunks at byte offsets 16g and 32+16g of row m.
DEVINL v16h load_afrag(const h16* A, int ld) {
  int lane = threadIdx.x & 31;
  int m = lane & 15, g = lane >> 4;
  const char* base = (const char*)(A + (size_t)m * ld) + (g << 4);
  HVec r;
  r.q[0] = *(const uint4*)(base);
  r.q[1] = *(const uint4*)(base + 32);
  return r.v;
}

// B-fragment (32x16 f16) from N-major storage BT[n*ld + k]:
// lane n = lane&15, g = lane>>4; VGPR v holds pair k = (v>>2)*16 + g*8 + (v&3)*2
// -> two 16-byte chunks at byte offsets 16g and 32+16g of row n.
DEVINL v16h load_bfrag(const h16* BT, int ld) {
  int lane = threadIdx.x & 31;
  int n = lane & 15, g = lane >> 4;
  const char* base = (const char*)(BT + (size_t)n * ld) + (g << 4);
  HVec r;
  r.q[0] = *(const uint4*)(base);
  r.q[1] = *(const uint4*)(base + 32);
  return r.v;
}

// Register-blocked GEMM: one wave computes 16 rows x 64 cols (4 acc tiles),
// reusing the A fragment across 4 B tiles -> 1.25 fragment loads per WMMA
// and 4 independent accumulators to cover WMMA->WMMA RAW latency.
template <int KDIM>
DEVINL void gemm_tile4(const h16* __restrict__ A, const h16* __restrict__ BT,
                       v8f acc[4]) {
  __builtin_prefetch((const char*)A + 2048, 0, 1);   // global_prefetch_b8
  __builtin_prefetch((const char*)BT + 2048, 0, 1);
#pragma unroll
  for (int kk = 0; kk < KDIM; kk += 32) {
    v16h a = load_afrag(A + kk, KDIM);
#pragma unroll
    for (int c = 0; c < 4; ++c) {
      v16h b = load_bfrag(BT + (size_t)c * 16 * KDIM + kk, KDIM);
      acc[c] = __builtin_amdgcn_wmma_f32_16x16x32_f16(false, a, false, b,
                                                      (short)0, acc[c],
                                                      false, false);
    }
  }
}

#if USE_TDM
// Tensor Data Mover: 1-D byte tile global -> LDS (ISA 08_async_tensor #8).
// Group0: count=1 | lds_addr | global_addr[56:0] | type=2.
// Group1: data_size=1B, tensor_dim0=tile_dim0=nbytes, tensor_dim1=1, 1-D.
DEVINL void tdm_load_bytes(const void* gsrc, void* ldst, unsigned nbytes) {
  uint64_t ga = (uint64_t)(uintptr_t)gsrc;
  unsigned lo = (unsigned)(uintptr_t)ldst;   // flat LDS addr low 32 = offset
  u32x4 g0;
  g0[0] = 1u;                                            // count=1 (user)
  g0[1] = lo;                                            // lds_addr
  g0[2] = (unsigned)ga;                                  // global_addr lo
  g0[3] = (unsigned)((ga >> 32) & 0x01FFFFFFu) | (2u << 30); // hi + type=2
  i32x8 g1;
  g1[0] = 0;                                             // wg_mask, ds=1B
  g1[1] = (int)((nbytes & 0xFFFFu) << 16);               // tensor_dim0 lo
  g1[2] = (int)((nbytes >> 16) & 0xFFFFu) | (1 << 16);   // td0 hi, td1=1
  g1[3] = (int)((nbytes & 0xFFFFu) << 16);               // tile_dim0
  g1[4] = 0;                                             // tile_dim1/2 = 0
  g1[5] = (int)nbytes;                                   // td0_stride lo
  g1[6] = 0;
  g1[7] = 0;
  i32x4 gz = {0, 0, 0, 0};
#if __clang_major__ >= 23
  i32x8 gz8 = {0, 0, 0, 0, 0, 0, 0, 0};
  __builtin_amdgcn_tensor_load_to_lds(g0, g1, gz, gz, gz8, 0);
#else
  __builtin_amdgcn_tensor_load_to_lds(g0, g1, gz, gz, 0);
#endif
}
#endif

// ---------------------------------------------------------------------------
// Problem constants: B=2, T=8, H=W=96, DIM=192, win=(2,8,8), shift=(1,4,4)
// windows: 4 x 12 x 12 per batch -> 576; total 1152; 128 tokens/window.
// ---------------------------------------------------------------------------

// weight transpose fp32 (K,N) -> f16 (N,K)
__global__ void k_transpose(const float* __restrict__ src, h16* __restrict__ dst,
                            int K, int N) {
  int idx = blockIdx.x * 256 + threadIdx.x;
  if (idx >= K * N) return;
  int k = idx / N, n = idx - k * N;
  dst[(size_t)n * K + k] = (h16)src[idx];
}

// LN1 + shift-roll + window partition gather -> Xw f16 [1152*128][192]
__global__ __launch_bounds__(192) void k_ln1_window(
    const float* __restrict__ x, const float* __restrict__ gw,
    const float* __restrict__ bw, h16* __restrict__ xw) {
  __shared__ float red1[6], red2[6];
  int m = blockIdx.x;
  int win = m >> 7, tok = m & 127;
  int b = win / 576, wi = win - b * 576;
  int tT = wi / 144; int rem = wi - tT * 144;
  int hH = rem / 12, wW = rem - hH * 12;
  int dt = tok >> 6, dh = (tok >> 3) & 7, dw = tok & 7;
  int t = (tT * 2 + dt + 1) & 7;                 // roll(-1) gather
  int h = hH * 8 + dh + 4; if (h >= 96) h -= 96; // roll(-4)
  int w = wW * 8 + dw + 4; if (w >= 96) w -= 96;
  const float* row = x + ((((size_t)b * 8 + t) * 96 + h) * 96 + w) * 192;
  int c = threadIdx.x;
  float v = row[c];
  float s = v;
#pragma unroll
  for (int off = 16; off > 0; off >>= 1) s += __shfl_xor(s, off, 32);
  if ((c & 31) == 0) red1[c >> 5] = s;
  __syncthreads();
  float mean = (red1[0] + red1[1] + red1[2] + red1[3] + red1[4] + red1[5]) * (1.0f / 192.0f);
  float d = v - mean;
  float q = d * d;
#pragma unroll
  for (int off = 16; off > 0; off >>= 1) q += __shfl_xor(q, off, 32);
  if ((c & 31) == 0) red2[c >> 5] = q;
  __syncthreads();
  float var = (red2[0] + red2[1] + red2[2] + red2[3] + red2[4] + red2[5]) * (1.0f / 192.0f);
  float rn = rsqrtf(var + 1e-5f);
  xw[(size_t)m * 192 + c] = (h16)(d * rn * gw[c] + bw[c]);
}

// plain LN over x2 rows -> Yn f16
__global__ __launch_bounds__(192) void k_ln2(
    const float* __restrict__ x2, const float* __restrict__ gw,
    const float* __restrict__ bw, h16* __restrict__ yn) {
  __shared__ float red1[6], red2[6];
  const float* row = x2 + (size_t)blockIdx.x * 192;
  int c = threadIdx.x;
  float v = row[c];
  float s = v;
#pragma unroll
  for (int off = 16; off > 0; off >>= 1) s += __shfl_xor(s, off, 32);
  if ((c & 31) == 0) red1[c >> 5] = s;
  __syncthreads();
  float mean = (red1[0] + red1[1] + red1[2] + red1[3] + red1[4] + red1[5]) * (1.0f / 192.0f);
  float d = v - mean;
  float q = d * d;
#pragma unroll
  for (int off = 16; off > 0; off >>= 1) q += __shfl_xor(q, off, 32);
  if ((c & 31) == 0) red2[c >> 5] = q;
  __syncthreads();
  float var = (red2[0] + red2[1] + red2[2] + red2[3] + red2[4] + red2[5]) * (1.0f / 192.0f);
  float rn = rsqrtf(var + 1e-5f);
  yn[(size_t)blockIdx.x * 192 + c] = (h16)(d * rn * gw[c] + bw[c]);
}

// QKV GEMM: M=147456, K=192, N=576. Wave = 16 rows x 64 cols.
// Epilogue splits Q(scaled)/K/V(transposed to dim-major).
__global__ __launch_bounds__(128) void k_qkv(
    const h16* __restrict__ Xw, const h16* __restrict__ WT,
    const float* __restrict__ bias, h16* __restrict__ Qb,
    h16* __restrict__ Kb, h16* __restrict__ Vt) {
  int wave = threadIdx.x >> 5;
  int rowTile = blockIdx.x * 4 + wave;
  int colBase = blockIdx.y * 64;
  v8f acc[4] = {};
  gemm_tile4<192>(Xw + (size_t)rowTile * 16 * 192,
                  WT + (size_t)colBase * 192, acc);
  int lane = threadIdx.x & 31;
  int nloc = lane & 15, g = lane >> 4;
#pragma unroll
  for (int c = 0; c < 4; ++c) {
    int col = colBase + c * 16 + nloc;
    int which = col / 192;
    int jj = col - which * 192;
    int head = jj >> 5, dd = jj & 31;
    float bv = bias[col];
#pragma unroll
    for (int r = 0; r < 8; ++r) {
      int m = rowTile * 16 + r + (g << 3);
      int win = m >> 7, tok = m & 127;
      float val = acc[c][r] + bv;
      size_t hb = (size_t)win * 6 + head;
      if (which == 0)
        Qb[(hb * 128 + tok) * 32 + dd] = (h16)(val * 0.17677669529663687f);
      else if (which == 1)
        Kb[(hb * 128 + tok) * 32 + dd] = (h16)val;
      else
        Vt[(hb * 32 + dd) * 128 + tok] = (h16)val;   // V stored dim-major
    }
  }
}

DEVINL int region3d(int ts, int hs, int ws) {
  int rT = ts < 6 ? 0 : (ts < 7 ? 1 : 2);
  int rH = hs < 88 ? 0 : (hs < 92 ? 1 : 2);
  int rW = ws < 88 ? 0 : (ws < 92 ? 1 : 2);
  return rT * 9 + rH * 3 + rW;
}

// Attention: one block = (window, head); 8 waves x 16 rows of the 128x128 S.
// K (128x32) and V (32x128, dim-major) staged once per block in LDS via the
// Tensor Data Mover (fallback: vectorized copy); P staged per-wave in LDS.
__global__ __launch_bounds__(256) void k_attn(
    const h16* __restrict__ Qb, const h16* __restrict__ Kb,
    const h16* __restrict__ Vt, const float* __restrict__ rpb,
    h16* __restrict__ O) {
  __shared__ h16 Klds[128 * 32];      // 8 KB
  __shared__ h16 Vlds[32 * 128];      // 8 KB
  __shared__ h16 Plds[8 * 16 * 128];  // 32 KB
  int win = blockIdx.x / 6, head = blockIdx.x - win * 6;
  int wave = threadIdx.x >> 5, lane = threadIdx.x & 31;
  int nloc = lane & 15, g = lane >> 4;
  size_t hb = (size_t)win * 6 + head;
  const h16* Q  = Qb + (hb << 12);
  const h16* Kp = Kb + (hb << 12);
  const h16* Vp = Vt + (hb << 12);
  int rowBase = wave * 16;

#if USE_TDM
  if (threadIdx.x < 32) {                 // wave 0 drives the TDM
    tdm_load_bytes(Kp, Klds, 128 * 32 * 2);
    tdm_load_bytes(Vp, Vlds, 32 * 128 * 2);
    __builtin_amdgcn_s_wait_tensorcnt(0); // data in LDS before barrier
  }
#else
  {
    const uint4* gk = (const uint4*)Kp;
    uint4* sk = (uint4*)Klds;
    sk[threadIdx.x]       = gk[threadIdx.x];
    sk[threadIdx.x + 256] = gk[threadIdx.x + 256];
    const uint4* gv = (const uint4*)Vp;
    uint4* sv = (uint4*)Vlds;
    sv[threadIdx.x]       = gv[threadIdx.x];
    sv[threadIdx.x + 256] = gv[threadIdx.x + 256];
  }
#endif
  __syncthreads();

  // S = Q K^T : A = Q rows (16x32), B = K tokens as columns from LDS
  v16h qa = load_afrag(Q + (size_t)rowBase * 32, 32);
  v8f S[8];
#pragma unroll
  for (int ct = 0; ct < 8; ++ct) {
    v16h kb = load_bfrag(Klds + (size_t)ct * 16 * 32, 32);
    v8f z = {};
    S[ct] = __builtin_amdgcn_wmma_f32_16x16x32_f16(false, qa, false, kb,
                                                   (short)0, z, false, false);
  }

  // relative position bias + shift mask (recomputed analytically)
  int wi = win % 576;
  int tT = wi / 144; int rem = wi - tT * 144;
  int hH = rem / 12, wW = rem - hH * 12;
#pragma unroll
  for (int ct = 0; ct < 8; ++ct) {
    int j = ct * 16 + nloc;
    int djt = j >> 6, djh = (j >> 3) & 7, djw = j & 7;
    int rj = region3d(tT * 2 + djt, hH * 8 + djh, wW * 8 + djw);
#pragma unroll
    for (int r = 0; r < 8; ++r) {
      int i = rowBase + r + (g << 3);
      int dit = i >> 6, dih = (i >> 3) & 7, diw = i & 7;
      int ri = region3d(tT * 2 + dit, hH * 8 + dih, wW * 8 + diw);
      int ridx = (dit - djt + 1) * 225 + (dih - djh + 7) * 15 + (diw - djw + 7);
      float bias = rpb[ridx * 6 + head];
      if (ri != rj) bias -= 100.0f;
      S[ct][r] += bias;
    }
  }

  // softmax: row r+8g lives in the 16 lanes sharing g (xor<=8 stays in half)
#pragma unroll
  for (int r = 0; r < 8; ++r) {
    float mx = S[0][r];
#pragma unroll
    for (int ct = 1; ct < 8; ++ct) mx = fmaxf(mx, S[ct][r]);
#pragma unroll
    for (int off = 1; off < 16; off <<= 1) mx = fmaxf(mx, __shfl_xor(mx, off, 32));
    float sum = 0.0f;
#pragma unroll
    for (int ct = 0; ct < 8; ++ct) {
      float e = __expf(S[ct][r] - mx);
      S[ct][r] = e;
      sum += e;
    }
#pragma unroll
    for (int off = 1; off < 16; off <<= 1) sum += __shfl_xor(sum, off, 32);
    float inv = 1.0f / sum;
#pragma unroll
    for (int ct = 0; ct < 8; ++ct) S[ct][r] *= inv;
  }

  // stage P in LDS row-major [16][128] per wave, reload as A-fragments
  h16* Pw = &Plds[wave * 16 * 128];
#pragma unroll
  for (int ct = 0; ct < 8; ++ct)
#pragma unroll
    for (int r = 0; r < 8; ++r)
      Pw[(r + (g << 3)) * 128 + ct * 16 + nloc] = (h16)S[ct][r];
  __syncthreads();

  // O = P V : B from dim-major V in LDS (pairs over token contiguous)
#pragma unroll
  for (int nt = 0; nt < 2; ++nt) {
    v8f acc = {};
#pragma unroll
    for (int kc = 0; kc < 4; ++kc) {
      v16h pa = load_afrag(Pw + kc * 32, 128);
      v16h vb = load_bfrag(Vlds + (size_t)nt * 16 * 128 + kc * 32, 128);
      acc = __builtin_amdgcn_wmma_f32_16x16x32_f16(false, pa, false, vb,
                                                   (short)0, acc, false, false);
    }
#pragma unroll
    for (int r = 0; r < 8; ++r) {
      int tok = rowBase + r + (g << 3);
      int ch = head * 32 + nt * 16 + nloc;
      O[((size_t)win * 128 + tok) * 192 + ch] = (h16)acc[r];
    }
  }
}

// proj GEMM + window reverse + un-shift scatter + residual -> x2 (f32)
__global__ __launch_bounds__(128) void k_proj(
    const h16* __restrict__ O, const h16* __restrict__ WT,
    const float* __restrict__ bias, const float* __restrict__ x,
    float* __restrict__ x2) {
  int wave = threadIdx.x >> 5;
  int rowTile = blockIdx.x * 4 + wave;
  int colBase = blockIdx.y * 64;
  v8f acc[4] = {};
  gemm_tile4<192>(O + (size_t)rowTile * 16 * 192,
                  WT + (size_t)colBase * 192, acc);
  int lane = threadIdx.x & 31;
  int nloc = lane & 15, g = lane >> 4;
  float bv[4];
#pragma unroll
  for (int c = 0; c < 4; ++c) bv[c] = bias[colBase + c * 16 + nloc];
#pragma unroll
  for (int r = 0; r < 8; ++r) {
    int m = rowTile * 16 + r + (g << 3);
    int win = m >> 7, tok = m & 127;
    int b = win / 576, wi = win - b * 576;
    int tT = wi / 144; int rem = wi - tT * 144;
    int hH = rem / 12, wW = rem - hH * 12;
    int dt = tok >> 6, dh = (tok >> 3) & 7, dw = tok & 7;
    int t = (tT * 2 + dt + 1) & 7;
    int h = hH * 8 + dh + 4; if (h >= 96) h -= 96;
    int w = wW * 8 + dw + 4; if (w >= 96) w -= 96;
    size_t base = ((((size_t)b * 8 + t) * 96 + h) * 96 + w) * 192;
#pragma unroll
    for (int c = 0; c < 4; ++c) {
      int col = colBase + c * 16 + nloc;
      x2[base + col] = x[base + col] + acc[c][r] + bv[c];
    }
  }
}

// fc1 GEMM + exact GELU -> Hh f16 [M][384]
__global__ __launch_bounds__(128) void k_fc1(
    const h16* __restrict__ Yn, const h16* __restrict__ WT,
    const float* __restrict__ bias, h16* __restrict__ Hh) {
  int wave = threadIdx.x >> 5;
  int rowTile = blockIdx.x * 4 + wave;
  int colBase = blockIdx.y * 64;
  v8f acc[4] = {};
  gemm_tile4<192>(Yn + (size_t)rowTile * 16 * 192,
                  WT + (size_t)colBase * 192, acc);
  int lane = threadIdx.x & 31;
  int nloc = lane & 15, g = lane >> 4;
#pragma unroll
  for (int c = 0; c < 4; ++c) {
    int col = colBase + c * 16 + nloc;
    float bv = bias[col];
#pragma unroll
    for (int r = 0; r < 8; ++r) {
      int m = rowTile * 16 + r + (g << 3);
      float u = acc[c][r] + bv;
      float ge = 0.5f * u * (1.0f + erff(u * 0.70710678118654752f));
      Hh[(size_t)m * 384 + col] = (h16)ge;
    }
  }
}

// fc2 GEMM + bias + residual -> out f32
__global__ __launch_bounds__(128) void k_fc2(
    const h16* __restrict__ Hh, const h16* __restrict__ WT,
    const float* __restrict__ bias, const float* __restrict__ x2,
    float* __restrict__ out) {
  int wave = threadIdx.x >> 5;
  int rowTile = blockIdx.x * 4 + wave;
  int colBase = blockIdx.y * 64;
  v8f acc[4] = {};
  gemm_tile4<384>(Hh + (size_t)rowTile * 16 * 384,
                  WT + (size_t)colBase * 384, acc);
  int lane = threadIdx.x & 31;
  int nloc = lane & 15, g = lane >> 4;
  float bv[4];
#pragma unroll
  for (int c = 0; c < 4; ++c) bv[c] = bias[colBase + c * 16 + nloc];
#pragma unroll
  for (int r = 0; r < 8; ++r) {
    size_t m = (size_t)(rowTile * 16 + r + (g << 3));
#pragma unroll
    for (int c = 0; c < 4; ++c) {
      int col = colBase + c * 16 + nloc;
      out[m * 192 + col] = x2[m * 192 + col] + acc[c][r] + bv[c];
    }
  }
}

// ---------------------------------------------------------------------------
extern "C" void kernel_launch(void* const* d_in, const int* in_sizes, int n_in,
                              void* d_out, int out_size, void* d_ws, size_t ws_size,
                              hipStream_t stream) {
  (void)in_sizes; (void)n_in; (void)out_size; (void)ws_size;
  const float* x      = (const float*)d_in[0];
  const float* g1     = (const float*)d_in[1];
  const float* b1     = (const float*)d_in[2];
  const float* g2     = (const float*)d_in[3];
  const float* b2     = (const float*)d_in[4];
  const float* qkv_w  = (const float*)d_in[5];
  const float* qkv_b  = (const float*)d_in[6];
  const float* proj_w = (const float*)d_in[7];
  const float* proj_b = (const float*)d_in[8];
  const float* rpb    = (const float*)d_in[9];
  const float* fc1_w  = (const float*)d_in[10];
  const float* fc1_b  = (const float*)d_in[11];
  const float* fc2_w  = (const float*)d_in[12];
  const float* fc2_b  = (const float*)d_in[13];
  float* out = (float*)d_out;

  const size_t M = 147456;                 // 1152 windows * 128 tokens
  const size_t SZH = M * 192 * 2;          // one f16 activation buffer
  char* ws = (char*)d_ws;
  h16* Xw   = (h16*)(ws);                  // also aliases O (attn output)
  h16* Obuf = Xw;
  h16* Qb   = (h16*)(ws + SZH);
  h16* Kb   = (h16*)(ws + 2 * SZH);
  h16* Vt   = (h16*)(ws + 3 * SZH);
  float* x2 = (float*)(ws + 4 * SZH);
  char* wt  = ws + 4 * SZH + M * 192 * 4;
  h16* qkvT = (h16*)(wt);
  h16* projT = (h16*)(wt + (size_t)576 * 192 * 2);
  h16* fc1T = (h16*)(wt + (size_t)576 * 192 * 2 + (size_t)192 * 192 * 2);
  h16* fc2T = (h16*)(wt + (size_t)576 * 192 * 2 + (size_t)192 * 192 * 2 +
                     (size_t)384 * 192 * 2);
  h16* Yn = Qb;   // reuse Q after attention
  h16* Hh = Kb;   // reuse K+Vt (contiguous 113MB) after attention

  // weight prep (fp32 -> f16, N-major)
  k_transpose<<<(192 * 576 + 255) / 256, 256, 0, stream>>>(qkv_w, qkvT, 192, 576);
  k_transpose<<<(192 * 192 + 255) / 256, 256, 0, stream>>>(proj_w, projT, 192, 192);
  k_transpose<<<(192 * 384 + 255) / 256, 256, 0, stream>>>(fc1_w, fc1T, 192, 384);
  k_transpose<<<(384 * 192 + 255) / 256, 256, 0, stream>>>(fc2_w, fc2T, 384, 192);

  // LN1 + shifted-window gather
  k_ln1_window<<<(int)M, 192, 0, stream>>>(x, g1, b1, Xw);
  // QKV projection (blocks: 64 rows x 64 cols, 4 waves)
  k_qkv<<<dim3(M / 64, 9), 128, 0, stream>>>(Xw, qkvT, qkv_b, Qb, Kb, Vt);
  // windowed attention
  k_attn<<<1152 * 6, 256, 0, stream>>>(Qb, Kb, Vt, rpb, Obuf);
  // output projection + window reverse + residual
  k_proj<<<dim3(M / 64, 3), 128, 0, stream>>>(Obuf, projT, proj_b, x, x2);
  // LN2 + MLP + residual
  k_ln2<<<(int)M, 192, 0, stream>>>(x2, g2, b2, Yn);
  k_fc1<<<dim3(M / 64, 6), 128, 0, stream>>>(Yn, fc1T, fc1_b, Hh);
  k_fc2<<<dim3(M / 64, 3), 128, 0, stream>>>(Hh, fc2T, fc2_b, x2, out);
}
